// NSF_CL_9698036154975
// MI455X (gfx1250) — matrix operand order repaired and checked by reading, compiled, not run on hardware
//
#include <hip/hip_runtime.h>

typedef __attribute__((ext_vector_type(16))) _Float16 v16h;
typedef __attribute__((ext_vector_type(8)))  _Float16 v8h;
typedef __attribute__((ext_vector_type(8)))  float    v8f;

// ---------------------------------------------------------------------------
// WMMA helpers (CDNA5 wave32, V_WMMA_F32_16X16X32_F16)
// ---------------------------------------------------------------------------
__device__ __forceinline__ v8f wmma_f16(v16h a, v16h b, v8f c) {
    return __builtin_amdgcn_wmma_f32_16x16x32_f16(
        /*neg_a=*/false, a, /*neg_b=*/false, b,
        /*c_mod=*/(short)0, c, /*reuse_a=*/false, /*reuse_b=*/false);
}

// A fragment (16x32 f16, MxK): lane m = l&15; halves 0..7 -> K = hb*8 + 0..7,
// halves 8..15 -> K = 16 + hb*8 + 0..7, where hb = l>>4.
__device__ __forceinline__ v16h load_a_frag_f16(const _Float16* __restrict__ A,
                                                int lda, int rowBase, int kb, int lane) {
    int m = lane & 15, hb = lane >> 4;
    const _Float16* base = A + (size_t)(rowBase + m) * lda + kb + hb * 8;
    v8h lo = *(const v8h*)(base);
    v8h hi = *(const v8h*)(base + 16);
    return __builtin_shufflevector(lo, hi, 0,1,2,3,4,5,6,7,8,9,10,11,12,13,14,15);
}

__device__ __forceinline__ v16h load_a_frag_f32(const float* __restrict__ A,
                                                int lda, int rowBase, int kb, int lane) {
    int m = lane & 15, hb = lane >> 4;
    const float* base = A + (size_t)(rowBase + m) * lda + kb + hb * 8;
    v16h a;
#pragma unroll
    for (int i = 0; i < 8; ++i) a[i] = (_Float16)base[i];
#pragma unroll
    for (int i = 0; i < 8; ++i) a[8 + i] = (_Float16)base[16 + i];
    return a;
}

// A fragment sourced from LDS (f16, row stride 512)
__device__ __forceinline__ v16h load_a_frag_lds(const _Float16* hsh, int kb, int lane) {
    int m = lane & 15, hb = lane >> 4;
    const _Float16* base = hsh + m * 512 + kb + hb * 8;
    v8h lo = *(const v8h*)(base);
    v8h hi = *(const v8h*)(base + 16);
    return __builtin_shufflevector(lo, hi, 0,1,2,3,4,5,6,7,8,9,10,11,12,13,14,15);
}

// B fragment (32x16 f16, KxN): W stored transposed as [Ncols, K] row-major.
// lane n = l&15; halves 0..7 -> K = hb*8 + 0..7, halves 8..15 -> K = 16 + hb*8 + 0..7.
__device__ __forceinline__ v16h load_b_frag(const _Float16* __restrict__ Wt,
                                            int ldk, int colBase, int kb, int lane) {
    const _Float16* base = Wt + (size_t)(colBase + (lane & 15)) * ldk + kb + ((lane >> 4) << 3);
    v8h lo = *(const v8h*)(base);
    v8h hi = *(const v8h*)(base + 16);
    return __builtin_shufflevector(lo, hi, 0,1,2,3,4,5,6,7,8,9,10,11,12,13,14,15);
}

// ---------------------------------------------------------------------------
// RQS spline element (faithful to reference, fp32)
// ---------------------------------------------------------------------------
__device__ __forceinline__ float softplusf(float v) {
    return (v > 20.0f) ? v : log1pf(expf(v));
}

__device__ __forceinline__ void softmax5_scaled(const float* in, float* out, float scale) {
    float m = in[0];
#pragma unroll
    for (int i = 1; i < 5; ++i) m = fmaxf(m, in[i]);
    float e[5]; float s = 0.0f;
#pragma unroll
    for (int i = 0; i < 5; ++i) { e[i] = expf(in[i] - m); s += e[i]; }
    float inv = scale / s;
#pragma unroll
    for (int i = 0; i < 5; ++i) out[i] = e[i] * inv;
}

__device__ __forceinline__ void rqs_elem(const float* p, float x_in,
                                         float* y_out, float* ld_out) {
    const float Bc = 3.0f, MINBW = 0.001f, MINBH = 0.001f, MIND = 0.001f;
    // _cond: Wu = 2B*softmax, Hu = 2B*softmax, Du = softplus
    float Wu[5], Hu[5], Du[4];
    softmax5_scaled(p,     Wu, 2.0f * Bc);
    softmax5_scaled(p + 5, Hu, 2.0f * Bc);
#pragma unroll
    for (int i = 0; i < 4; ++i) Du[i] = softplusf(p[10 + i]);

    // widths / knots x
    float wd[5]; softmax5_scaled(Wu, wd, 1.0f);
#pragma unroll
    for (int i = 0; i < 5; ++i) wd[i] = MINBW + (1.0f - MINBW * 5.0f) * wd[i];
    float cw[6]; cw[0] = -Bc;
    {
        float c = 0.0f;
#pragma unroll
        for (int i = 0; i < 5; ++i) { c += wd[i]; cw[i + 1] = 2.0f * Bc * c - Bc; }
        cw[5] = Bc;
    }
#pragma unroll
    for (int i = 0; i < 5; ++i) wd[i] = cw[i + 1] - cw[i];

    // heights / knots y
    float ht[5]; softmax5_scaled(Hu, ht, 1.0f);
#pragma unroll
    for (int i = 0; i < 5; ++i) ht[i] = MINBH + (1.0f - MINBH * 5.0f) * ht[i];
    float ch[6]; ch[0] = -Bc;
    {
        float c = 0.0f;
#pragma unroll
        for (int i = 0; i < 5; ++i) { c += ht[i]; ch[i + 1] = 2.0f * Bc * c - Bc; }
        ch[5] = Bc;
    }
#pragma unroll
    for (int i = 0; i < 5; ++i) ht[i] = ch[i + 1] - ch[i];

    // derivatives (padded with const so softplus(const)+MIN_D == 1)
    float constc = logf(expf(1.0f - MIND) - 1.0f);
    float dv[6];
    dv[0] = MIND + softplusf(constc);
#pragma unroll
    for (int i = 0; i < 4; ++i) dv[i + 1] = MIND + softplusf(Du[i]);
    dv[5] = MIND + softplusf(constc);

    // bin search
    float xc = fminf(fmaxf(x_in, -Bc), Bc);
    int cnt = 0;
#pragma unroll
    for (int i = 0; i < 6; ++i) cnt += (xc >= cw[i]) ? 1 : 0;
    int idx = cnt - 1; idx = idx < 0 ? 0 : (idx > 4 ? 4 : idx);

    float in_cw = cw[0], in_w = wd[0], in_ch = ch[0], in_h = ht[0];
    float dk = dv[0], dk1 = dv[1];
#pragma unroll
    for (int i = 1; i < 5; ++i) {
        bool m = (idx == i);
        in_cw = m ? cw[i] : in_cw;
        in_w  = m ? wd[i] : in_w;
        in_ch = m ? ch[i] : in_ch;
        in_h  = m ? ht[i] : in_h;
        dk    = m ? dv[i] : dk;
        dk1   = m ? dv[i + 1] : dk1;
    }

    float delta = in_h / in_w;
    float theta = (xc - in_cw) / in_w;
    float om    = 1.0f - theta;
    float t1m   = theta * om;
    float num   = in_h * (delta * theta * theta + dk * t1m);
    float den   = delta + (dk + dk1 - 2.0f * delta) * t1m;
    float y     = in_ch + num / den;
    float dnum  = delta * delta * (dk1 * theta * theta + 2.0f * delta * t1m + dk * om * om);
    float ld    = logf(dnum) - 2.0f * logf(den);

    bool inside = (x_in >= -Bc) && (x_in <= Bc);
    *y_out  = inside ? y : x_in;
    *ld_out = inside ? ld : 0.0f;
}

// ---------------------------------------------------------------------------
// Weight prep: transpose [K,N] fp32 -> [N,K] f16
// ---------------------------------------------------------------------------
__global__ void transpose_to_f16(const float* __restrict__ src, _Float16* __restrict__ dst,
                                 int K, int N) {
    int i = blockIdx.x * blockDim.x + threadIdx.x;
    int total = K * N;
    if (i < total) {
        int k = i / N, n = i - k * N;
        dst[(size_t)n * K + k] = (_Float16)src[i];
    }
}

// ---------------------------------------------------------------------------
// Fused coupling kernel: MLP (2 GEMMs, params tile kept in LDS) + RQS spline.
// Block = 256 threads (8 waves), owns 16 rows.
// ---------------------------------------------------------------------------
__global__ __launch_bounds__(256, 1)
void nsf_coupling_kernel(const float* __restrict__ x,          // [N,256]
                         const _Float16* __restrict__ a_f16,   // [N,128] or null
                         int a_is_f16,
                         const _Float16* __restrict__ w0t,     // [512,128] f16
                         const float* __restrict__ b0,         // [512]
                         const _Float16* __restrict__ w1t,     // [1792,512] f16
                         const float* __restrict__ b1,         // [1792]
                         int spline_col,                       // x column of spline input
                         float* __restrict__ z,                // [N,256] output
                         int z_col,
                         _Float16* __restrict__ xnext,         // [N,128] f16 or null
                         float* __restrict__ ld_buf,
                         float* __restrict__ ld_out,           // null on first coupling
                         int mode)                             // 0: write ld_buf, 1: final
{
    extern __shared__ char smem[];
    _Float16* hsh = (_Float16*)smem;                             // 16*512 f16  (16 KB)
    float*    psh = (float*)(smem + 16 * 512 * 2);               // 16*1792 f32 (112 KB)
    float*    red = (float*)(smem + 16 * 512 * 2 + 16 * 1792 * 4); // 256 f32

    const int rowBase = blockIdx.x * 16;
    const int lane = threadIdx.x & 31;
    const int wv   = threadIdx.x >> 5;   // 0..7
    const int nsel = lane & 15;
    const int hb   = lane >> 4;

    // ---------------- layer 1: h = tanh(x_half @ w0 + b0), K=128, 512 cols
    {
        v8f acc[4];
#pragma unroll
        for (int j = 0; j < 4; ++j)
#pragma unroll
            for (int r = 0; r < 8; ++r) acc[j][r] = 0.0f;

#pragma unroll
        for (int kb = 0; kb < 128; kb += 32) {
            v16h a = a_is_f16 ? load_a_frag_f16(a_f16, 128, rowBase, kb, lane)
                              : load_a_frag_f32(x, 256, rowBase, kb, lane);
#pragma unroll
            for (int j = 0; j < 4; ++j) {
                int colBase = wv * 64 + j * 16;
                v16h b = load_b_frag(w0t, 128, colBase, kb, lane);
                acc[j] = wmma_f16(a, b, acc[j]);
            }
        }
#pragma unroll
        for (int j = 0; j < 4; ++j) {
            int col = wv * 64 + j * 16 + nsel;
            float bias = b0[col];
#pragma unroll
            for (int r = 0; r < 8; ++r) {
                int m = r + 8 * hb;
                hsh[m * 512 + col] = (_Float16)tanhf(acc[j][r] + bias);
            }
        }
    }
    __syncthreads();

    // ---------------- layer 2: p = h @ w1 + b1, K=512, 1792 cols (wave: 14 tiles)
    {
        // Hoist the wave's entire K=512 A-strip into registers (16 frags = 128 VGPRs):
        // reused across all 14 column tiles, so the inner loop is pure B-load + WMMA.
        v16h afrag[16];
#pragma unroll
        for (int kk = 0; kk < 16; ++kk)
            afrag[kk] = load_a_frag_lds(hsh, kk * 32, lane);

        for (int t = wv; t < 112; t += 8) {
            int colBase = t * 16;
            v8f acc;
#pragma unroll
            for (int r = 0; r < 8; ++r) acc[r] = 0.0f;
#pragma unroll
            for (int kk = 0; kk < 16; ++kk) {
                v16h b = load_b_frag(w1t, 512, colBase, kk * 32, lane);
                acc = wmma_f16(afrag[kk], b, acc);
            }
            int col = colBase + nsel;
            float bias = b1[col];
#pragma unroll
            for (int r = 0; r < 8; ++r)
                psh[(r + 8 * hb) * 1792 + col] = acc[r] + bias;
        }
    }
    __syncthreads();

    // ---------------- spline: 16 threads per row, 8 dims per thread
    {
        int tt = threadIdx.x;
        int r = tt >> 4;
        int g = tt & 15;
        const size_t rowOff = (size_t)(rowBase + r) * 256;
        float lsum = 0.0f;
        for (int q = 0; q < 8; ++q) {
            int d = g * 8 + q;
            float x_in = x[rowOff + spline_col + d];
            float y, ld;
            rqs_elem(&psh[r * 1792 + d * 14], x_in, &y, &ld);
            z[rowOff + z_col + d] = y;
            if (xnext) xnext[(size_t)(rowBase + r) * 128 + d] = (_Float16)y;
            lsum += ld;
        }
        red[tt] = lsum;
        __syncthreads();
        if (g == 0) {
            float s = 0.0f;
#pragma unroll
            for (int i = 0; i < 16; ++i) s += red[r * 16 + i];
            int row = rowBase + r;
            if (mode == 0) ld_buf[row] = s;
            else           ld_out[row] = ld_buf[row] + s;
        }
    }
}

// ---------------------------------------------------------------------------
// Host launcher
// ---------------------------------------------------------------------------
extern "C" void kernel_launch(void* const* d_in, const int* in_sizes, int n_in,
                              void* d_out, int out_size, void* d_ws, size_t ws_size,
                              hipStream_t stream) {
    const float* x     = (const float*)d_in[0];
    const float* f0_w0 = (const float*)d_in[1];
    const float* f0_b0 = (const float*)d_in[2];
    const float* f0_w1 = (const float*)d_in[3];
    const float* f0_b1 = (const float*)d_in[4];
    const float* f1_w0 = (const float*)d_in[5];
    const float* f1_b0 = (const float*)d_in[6];
    const float* f1_w1 = (const float*)d_in[7];
    const float* f1_b1 = (const float*)d_in[8];

    const int N    = in_sizes[0] / 256;   // 32768
    const int HALF = 128, HID = 512, OUTD = 1792;

    // workspace layout (bytes)
    char* ws = (char*)d_ws;
    _Float16* w0t0   = (_Float16*)(ws);                      // 512*128*2   = 131072
    _Float16* w1t0   = (_Float16*)(ws + 131072);             // 1792*512*2  = 1835008
    _Float16* w0t1   = (_Float16*)(ws + 1966080);            // 131072
    _Float16* w1t1   = (_Float16*)(ws + 2097152);            // 1835008
    _Float16* x1next = (_Float16*)(ws + 3932160);            // N*128*2
    float*    ld_buf = (float*)(ws + 3932160 + (size_t)N * HALF * 2);

    float* z      = (float*)d_out;
    float* ld_out = z + (size_t)N * 256;

    // weight prep (fp32 [K,N] -> f16 [N,K])
    {
        int t1 = HALF * HID;   // 65536
        int t2 = HID * OUTD;   // 917504
        transpose_to_f16<<<(t1 + 255) / 256, 256, 0, stream>>>(f0_w0, w0t0, HALF, HID);
        transpose_to_f16<<<(t2 + 255) / 256, 256, 0, stream>>>(f0_w1, w1t0, HID, OUTD);
        transpose_to_f16<<<(t1 + 255) / 256, 256, 0, stream>>>(f1_w0, w0t1, HALF, HID);
        transpose_to_f16<<<(t2 + 255) / 256, 256, 0, stream>>>(f1_w1, w1t1, HID, OUTD);
    }

    size_t smem = (size_t)16 * 512 * 2 + (size_t)16 * 1792 * 4 + 256 * 4; // 132096 B

    // coupling 0: MLP(x0) -> spline on x1 (cols 128..255), emit f16 x1_new
    nsf_coupling_kernel<<<N / 16, 256, smem, stream>>>(
        x, nullptr, 0, w0t0, f0_b0, w1t0, f0_b1,
        /*spline_col=*/128, z, /*z_col=*/128, x1next, ld_buf, nullptr, /*mode=*/0);

    // coupling 1: MLP(x1_new) -> spline on x0 (cols 0..127), final log-det
    nsf_coupling_kernel<<<N / 16, 256, smem, stream>>>(
        x, x1next, 1, w0t1, f1_b0, w1t1, f1_b1,
        /*spline_col=*/0, z, /*z_col=*/0, nullptr, ld_buf, ld_out, /*mode=*/1);
}
